// myConv2d_26121991094702
// MI455X (gfx1250) — compile-verified, matrix operand order checked by reading, not run on hardware
//
#include <hip/hip_runtime.h>
#include <hip/hip_bf16.h>

// CDNA5 (gfx1250) implicit-GEMM conv2d, fp32 in/out.
// Math: split-precision 3x bf16 WMMA (AhiBhi + AhiBlo + AloBhi), f32 accumulate.
// GEMM view: M=256 (C_OUT), N=16*112*75=134400 (B*OH*OW), K=128*9=1152,
// reordered as k' = (kh*3+kw)*128 + ci so each 32-wide K-step has constant (kh,kw).
//
// Weights are pre-converted + pre-swizzled into the exact WMMA A-fragment layout
// in d_ws by a tiny pre-kernel (weights are 1.2 MB -> L2-resident); the main loop
// loads A operands with two global_load_b128 per lane, software-pipelined one
// K-step ahead in registers so the XDL pipe never waits on LOADcnt.

typedef __attribute__((ext_vector_type(16))) __bf16 v16bf;
typedef __attribute__((ext_vector_type(8)))  float  v8f;

namespace {
constexpr int CI = 128, IH = 224, IW = 224;
constexpr int CO = 256, OH = 112, OW = 75;
constexpr int NB = 16;
constexpr int KTOT   = CI * 9;          // 1152
constexpr int NTOT   = NB * OH * OW;    // 134400
constexpr int MT = 128, NT = 128, KT = 32;
constexpr int KSTEPS = KTOT / KT;       // 36
constexpr int LDST   = 48;              // padded B row stride (bf16) = 96B
constexpr int MSUBSG = CO / 16;         // 16 global M subtiles
// Pre-swizzled A: plane[((msubG*KSTEPS + kt)*32 + lane)*16 + e], e = frag elem.
constexpr size_t A_PLANE_ELEMS = (size_t)MSUBSG * KSTEPS * 32 * 16;   // 294912
constexpr size_t A_WS_BYTES    = A_PLANE_ELEMS * 2 /*bf16*/ * 2 /*hi+lo*/;
}

union FragBF { uint4 u[2]; v16bf v; };

// ---------------------------------------------------------------- pre-kernel
__global__ __launch_bounds__(256)
void preswizzle_weights(const float* __restrict__ wgt,
                        __bf16* __restrict__ hi, __bf16* __restrict__ lo)
{
    const int t = blockIdx.x * 256 + threadIdx.x;     // == output linear index
    if (t >= (int)A_PLANE_ELEMS) return;
    const int e     = t & 15;
    const int lane  = (t >> 4) & 31;
    const int kt    = (t >> 9) % KSTEPS;
    const int msubG = (t >> 9) / KSTEPS;
    const int halfL = lane >> 4, l15 = lane & 15;
    const int m     = msubG * 16 + l15;
    // ISA 16-bit A layout: element e holds k_local = (e&7) + 8*halfL + 16*(e>>3)
    const int kprime = kt * KT + ((e & 7) + 8 * halfL + 16 * (e >> 3));
    const int khw = kprime >> 7;          // kh*3+kw (0..8)
    const int ci  = kprime & 127;
    const float v = wgt[m * KTOT + ci * 9 + khw];
    const __bf16 h = (__bf16)v;
    hi[t] = h;
    lo[t] = (__bf16)(v - (float)h);
}

// ---------------------------------------------------------------- main kernel
template <bool PRE>
__global__ __launch_bounds__(256)
void conv3x3_s23_wmma_bf16x3(const float* __restrict__ x,
                             const float* __restrict__ wgt,
                             const float* __restrict__ bias,
                             const __bf16* __restrict__ wsAhi,
                             const __bf16* __restrict__ wsAlo,
                             float* __restrict__ out)
{
    // B tile only: [buf][hi/lo][n*LDST + k]
    __shared__ __align__(16) __bf16 sB[2][2][NT * LDST];

    const int tid   = threadIdx.x;
    const int mBase = blockIdx.y * MT;
    const int nBase = blockIdx.x * NT;

    // ---- B loader: 256 threads cover 128 n-rows x 32 k, 16 elems each
    const int nRow = tid >> 1;            // 0..127
    const int kCol = (tid & 1) * 16;      // 0 or 16

    const int nG  = nBase + nRow;
    const int bb  = nG / (OH * OW);
    const int rem = nG - bb * (OH * OW);
    const int oh  = rem / OW;
    const int ow  = rem - oh * OW;
    const int ih0 = oh * 2 - 1;           // pad=1, stride_h=2
    const int iw0 = ow * 3 - 1;           // pad=1, stride_w=3

    float rb[16];

    auto load_tile = [&](int kt) {
        // k' = khw*128 + ci  ->  khw = kt/4 (constant per step), ci = (kt&3)*32 + kCol + i
        const int khw = kt >> 2;
        const int kh  = khw / 3, kw = khw - kh * 3;
        const int ih  = ih0 + kh, iw = iw0 + kw;
        const int ci0 = (kt & 3) * 32 + kCol;
        const bool inb = ((unsigned)ih < (unsigned)IH) & ((unsigned)iw < (unsigned)IW);
        const float* px = x + (((size_t)bb * CI + ci0) * IH + ih) * IW + iw;
        #pragma unroll
        for (int i = 0; i < 16; ++i)
            rb[i] = inb ? px[(size_t)i * (IH * IW)] : 0.0f;
    };

    auto store_tile = [&](int buf) {
        __align__(16) __bf16 h[16], l[16];
        #pragma unroll
        for (int i = 0; i < 16; ++i) {
            const float v = rb[i];
            const __bf16 hh = (__bf16)v;
            h[i] = hh; l[i] = (__bf16)(v - (float)hh);
        }
        uint4* dh = (uint4*)&sB[buf][0][nRow * LDST + kCol];
        uint4* dl = (uint4*)&sB[buf][1][nRow * LDST + kCol];
        dh[0] = ((const uint4*)h)[0]; dh[1] = ((const uint4*)h)[1];
        dl[0] = ((const uint4*)l)[0]; dl[1] = ((const uint4*)l)[1];
    };

    // ---- wave mapping: 8 waves as 4(M strips of 32) x 2(N strips of 64)
    const int lane  = tid & 31;
    const int wv    = tid >> 5;
    const int wm    = wv >> 1;            // 0..3
    const int wn    = wv & 1;             // 0..1
    const int halfL = lane >> 4;
    const int l15   = lane & 15;
    const int msubG0 = (mBase / 16) + wm * 2;   // two M subtiles per wave

    auto loadA_pre = [&](const __bf16* plane, int msubG, int kt) -> v16bf {
        FragBF f;
        const uint4* p = (const uint4*)(plane + ((size_t)(msubG * KSTEPS + kt) * 32 + lane) * 16);
        f.u[0] = p[0]; f.u[1] = p[1];     // 32B contiguous per lane
        return f.v;
    };
    auto buildA = [&](int msubG, int kt, v16bf& aH, v16bf& aL) {
        FragBF H, L;
        __bf16* hp = (__bf16*)&H; __bf16* lp = (__bf16*)&L;
        const int m = msubG * 16 + l15;
        #pragma unroll
        for (int e = 0; e < 16; ++e) {
            const int kprime = kt * KT + ((e & 7) + 8 * halfL + 16 * (e >> 3));
            const int khw = kprime >> 7, ci = kprime & 127;
            const float v = wgt[m * KTOT + ci * 9 + khw];
            const __bf16 hh = (__bf16)v;
            hp[e] = hh; lp[e] = (__bf16)(v - (float)hh);
        }
        aH = H.v; aL = L.v;
    };
    auto fetchA = [&](int kt, v16bf (&aH)[2], v16bf (&aL)[2]) {
        if (PRE) {
            #pragma unroll
            for (int ms = 0; ms < 2; ++ms) {
                aH[ms] = loadA_pre(wsAhi, msubG0 + ms, kt);
                aL[ms] = loadA_pre(wsAlo, msubG0 + ms, kt);
            }
        } else {
            #pragma unroll
            for (int ms = 0; ms < 2; ++ms) buildA(msubG0 + ms, kt, aH[ms], aL[ms]);
        }
    };

    auto fragB = [&](const __bf16* base, int row) -> v16bf {
        FragBF f;
        const uint4* p = (const uint4*)(base + row * LDST);
        f.u[0] = p[halfL];
        f.u[1] = p[2 + halfL];
        return f.v;
    };

    v8f acc[2][4] = {};
    v16bf aHc[2], aLc[2];                 // current K-step A fragments
    v16bf aHn[2], aLn[2];                 // next K-step A fragments (pipelined)

    load_tile(0);
    store_tile(0);
    fetchA(0, aHc, aLc);                  // prologue: A frags for kt=0
    __syncthreads();

    for (int kt = 0; kt < KSTEPS; ++kt) {
        const int buf = kt & 1;
        if (kt + 1 < KSTEPS) {
            load_tile(kt + 1);            // global B gather for next step
            fetchA(kt + 1, aHn, aLn);     // global A frags for next step
        }

        #pragma unroll
        for (int ns = 0; ns < 4; ++ns) {
            const int bRow = wn * 64 + ns * 16 + l15;
            const v16bf bH = fragB(sB[buf][0], bRow);
            const v16bf bL = fragB(sB[buf][1], bRow);
            #pragma unroll
            for (int ms = 0; ms < 2; ++ms) {
                acc[ms][ns] = __builtin_amdgcn_wmma_f32_16x16x32_bf16(false, aHc[ms], false, bH, (short)0, acc[ms][ns], false, false);
                acc[ms][ns] = __builtin_amdgcn_wmma_f32_16x16x32_bf16(false, aHc[ms], false, bL, (short)0, acc[ms][ns], false, false);
                acc[ms][ns] = __builtin_amdgcn_wmma_f32_16x16x32_bf16(false, aLc[ms], false, bH, (short)0, acc[ms][ns], false, false);
            }
        }

        if (kt + 1 < KSTEPS) {
            store_tile((kt + 1) & 1);     // convert + stage next B tile
            #pragma unroll
            for (int ms = 0; ms < 2; ++ms) { aHc[ms] = aHn[ms]; aLc[ms] = aLn[ms]; }
        }
        __syncthreads();                  // one barrier per K-step
    }

    // ---- epilogue: C/D layout (VGPR j -> M = j + 8*halfL ; lane&15 -> N), + bias
    #pragma unroll
    for (int ns = 0; ns < 4; ++ns) {
        const int n  = nBase + wn * 64 + ns * 16 + l15;
        const int b0 = n / (OH * OW);
        const int r0 = n - b0 * (OH * OW);
        const int o0 = r0 / OW;
        const int w0 = r0 - o0 * OW;
        float* po = out + (((size_t)b0 * CO) * OH + o0) * OW + w0;
        #pragma unroll
        for (int ms = 0; ms < 2; ++ms) {
            const int mRow = mBase + wm * 32 + ms * 16 + 8 * halfL;
            #pragma unroll
            for (int j = 0; j < 8; ++j) {
                const int m = mRow + j;
                po[(size_t)m * (OH * OW)] = acc[ms][ns][j] + bias[m];
            }
        }
    }
}

// ---------------------------------------------------------------- launch
extern "C" void kernel_launch(void* const* d_in, const int* in_sizes, int n_in,
                              void* d_out, int out_size, void* d_ws, size_t ws_size,
                              hipStream_t stream) {
    (void)in_sizes; (void)n_in; (void)out_size;
    const float* x    = (const float*)d_in[0];
    const float* wgt  = (const float*)d_in[1];
    const float* bias = (const float*)d_in[2];
    float* out        = (float*)d_out;

    dim3 grid(NTOT / NT, CO / MT);   // 1050 x 2 workgroups
    dim3 block(256);                 // 8 wave32 waves

    if (ws_size >= A_WS_BYTES) {
        __bf16* hi = (__bf16*)d_ws;
        __bf16* lo = hi + A_PLANE_ELEMS;
        const int total  = (int)A_PLANE_ELEMS;
        preswizzle_weights<<<(total + 255) / 256, 256, 0, stream>>>(wgt, hi, lo);
        conv3x3_s23_wmma_bf16x3<true><<<grid, block, 0, stream>>>(x, wgt, bias, hi, lo, out);
    } else {
        conv3x3_s23_wmma_bf16x3<false><<<grid, block, 0, stream>>>(x, wgt, bias, nullptr, nullptr, out);
    }
}
